// SlidingWindowAttention_20736102105394
// MI455X (gfx1250) — matrix operand order repaired and checked by reading, not compiled, not run on hardware
//
#include <hip/hip_runtime.h>
#include <stdint.h>

#define T_SEQ   2048
#define DM      1024
#define NH      16
#define HD      64
#define WIN     256

typedef __attribute__((ext_vector_type(16))) __bf16 bf16x16;
typedef __attribute__((ext_vector_type(8)))  float  f32x8;

union Frag { bf16x16 v; unsigned u[8]; uint4 q4[2]; };

static __device__ inline unsigned short f2bf(float f){
  union { float f; unsigned u; } cv; cv.f = f;
  unsigned u = cv.u;
  unsigned r = u + 0x7FFFu + ((u >> 16) & 1u);   // round-to-nearest-even
  return (unsigned short)(r >> 16);
}
static __device__ inline unsigned pack2(float lo, float hi){
  return (unsigned)f2bf(lo) | ((unsigned)f2bf(hi) << 16);
}
static __device__ inline f32x8 zerov(){
  f32x8 z;
  #pragma unroll
  for(int i=0;i<8;i++) z[i]=0.0f;
  return z;
}
static __device__ inline f32x8 wmma_bf16(bf16x16 a, bf16x16 b, f32x8 c){
  return __builtin_amdgcn_wmma_f32_16x16x32_bf16(false, a, false, b, (short)0, c, false, false);
}

// ---------------------------------------------------------------------------
// Prep 1: x (f32) -> xb (bf16 row-major). 4M output dwords.
// ---------------------------------------------------------------------------
__global__ __launch_bounds__(256)
void cvt_x_kernel(const float2* __restrict__ x2, unsigned* __restrict__ xb)
{
  size_t i0 = (size_t)blockIdx.x*2048 + threadIdx.x;
  #pragma unroll
  for(int e=0;e<8;e++){
    size_t i = i0 + (size_t)e*256;
    float2 f = x2[i];
    xb[i] = pack2(f.x, f.y);
  }
}

// ---------------------------------------------------------------------------
// Prep 2: weight (K=1024 x N f32) -> bf16 pre-swizzled into WMMA B-fragment
// order: [kchunk(32)][ntile16][lane(32)][r(8)] dwords; dword r of lane L holds
// rows (kc*32 + (r>>2)*16 + (L>>4)*8 + 2(r&3), +1) at column nt*16 + (L&15).
// ---------------------------------------------------------------------------
__global__ __launch_bounds__(256)
void swz_w_kernel(const float* __restrict__ w, unsigned* __restrict__ wsw, int N, int ntiles)
{
  size_t o = (size_t)blockIdx.x*256 + threadIdx.x;
  int r    = (int)(o & 7);
  int lane = (int)((o >> 3) & 31);
  size_t tile = o >> 8;
  int nt = (int)(tile % ntiles);
  int kc = (int)(tile / ntiles);
  int n = nt*16 + (lane & 15);
  int k = kc*32 + ((r >> 2) << 4) + ((lane >> 4) << 3) + 2*(r & 3);
  wsw[o] = pack2(w[(size_t)k*N + n], w[(size_t)(k+1)*N + n]);
}

// ---------------------------------------------------------------------------
// Kernel 1: qkv = xb @ wqkv_sw. Wave = 32x64 tile (2 A frags, 8 accumulators):
// each B fragment feeds two WMMAs. Epilogue: q row-major [B][H][T][HD]
// (scaled by 1/8), k/v written directly in attention B-fragment order.
// ---------------------------------------------------------------------------
__global__ __launch_bounds__(256)
void qkv_gemm_kernel(const unsigned short* __restrict__ xb, const unsigned* __restrict__ wsw,
                     unsigned short* __restrict__ qb, unsigned short* __restrict__ kb,
                     unsigned short* __restrict__ vb)
{
  const int lane = threadIdx.x & 31;
  const int wv   = threadIdx.x >> 5;
  const int wid  = blockIdx.x*8 + wv;      // 12288 waves = 256 mtiles * 48 ntiles
  const int mt   = wid & 255;
  const int nt64 = wid >> 8;
  const int mbase = mt*32;
  const int nloc = lane & 15;
  const int half = lane >> 4;

  f32x8 cc0[4], cc1[4];
  #pragma unroll
  for(int i=0;i<4;i++){ cc0[i] = zerov(); cc1[i] = zerov(); }

  const unsigned short* arow0 = xb + (size_t)(mbase +      nloc)*DM;
  const unsigned short* arow1 = xb + (size_t)(mbase + 16 + nloc)*DM;

  #pragma unroll 1
  for(int kc=0; kc<32; kc++){
    Frag a0, a1;
    a0.q4[0] = *(const uint4*)(arow0 + kc*32 +      half*8);
    a0.q4[1] = *(const uint4*)(arow0 + kc*32 + 16 + half*8);
    a1.q4[0] = *(const uint4*)(arow1 + kc*32 +      half*8);
    a1.q4[1] = *(const uint4*)(arow1 + kc*32 + 16 + half*8);
    if(kc+1 < 32)
      __builtin_prefetch(wsw + ((size_t)(kc+1)*192 + nt64*4)*256 + lane*8, 0, 1);
    #pragma unroll
    for(int ns=0; ns<4; ns++){
      const unsigned* bp = wsw + ((size_t)kc*192 + nt64*4 + ns)*256 + lane*8;
      Frag b;
      b.q4[0] = *(const uint4*)(bp);
      b.q4[1] = *(const uint4*)(bp + 4);
      cc0[ns] = wmma_bf16(a0.v, b.v, cc0[ns]);
      cc1[ns] = wmma_bf16(a1.v, b.v, cc1[ns]);
    }
  }

  #pragma unroll
  for(int hs=0; hs<2; hs++){
    #pragma unroll
    for(int ns=0; ns<4; ns++){
      #pragma unroll
      for(int r=0;r<8;r++){
        int m   = mbase + hs*16 + r + 8*half;
        int col = nt64*64 + ns*16 + nloc;
        float val = hs ? cc1[ns][r] : cc0[ns][r];
        int sec = col >> 10;
        int c2  = col & (DM-1);
        int h   = c2 >> 6, d = c2 & (HD-1);
        int bi  = m >> 11, t = m & (T_SEQ-1);
        size_t bh = (size_t)bi*NH + h;
        if(sec == 0){
          qb[(bh*T_SEQ + t)*HD + d] = f2bf(val * 0.125f);
        } else if(sec == 1){
          // K^T fragments: K-dim = head dim d (32-chunk), N = key index t
          size_t frag = (bh*128 + (t >> 4))*2 + (d >> 5);
          int ls = (t & 15) | (((d >> 3) & 1) << 4);
          int rr = (((d >> 4) & 1) << 2) | ((d & 7) >> 1);
          kb[(frag*256 + ls*8 + rr)*2 + (d & 1)] = f2bf(val);
        } else {
          // V fragments: K-dim = key (32-chunk), N = d (16-subtile)
          size_t frag = (bh*64 + (t >> 5))*4 + (d >> 4);
          int ls = (d & 15) | (((t >> 3) & 1) << 4);
          int rr = (((t >> 4) & 1) << 2) | ((t & 7) >> 1);
          vb[(frag*256 + ls*8 + rr)*2 + (t & 1)] = f2bf(val);
        }
      }
    }
  }
}

// ---------------------------------------------------------------------------
// Kernel 2: sliding-window flash attention, one wave per (b, h, 16-row q tile).
// grid 1024, block 256 (8 independent waves). attn_out: [B][T][H][HD] bf16.
// ---------------------------------------------------------------------------
__global__ __launch_bounds__(256)
void swa_kernel(const unsigned short* __restrict__ q, const unsigned* __restrict__ ksw,
                const unsigned* __restrict__ vsw, unsigned short* __restrict__ ao)
{
  __shared__ unsigned short P[8][16*32];
  const int lane = threadIdx.x & 31;
  const int wv   = threadIdx.x >> 5;
  const int tile = blockIdx.x*8 + wv;       // 8192 tiles = B*H*(T/16)
  const int b    = tile >> 11;
  const int rem  = tile & 2047;
  const int h    = rem >> 7;
  const int t0   = (rem & 127) * 16;
  const size_t bh = (size_t)b*NH + h;

  const int nloc = lane & 15;
  const int half = lane >> 4;

  const unsigned short* qbase = q + (bh*T_SEQ + t0)*HD;
  const unsigned* kbh = ksw + bh*128*2*256;   // [ktile16][dchunk2][lane][r]
  const unsigned* vbh = vsw + bh*64*4*256;    // [kchunk32][dtile4][lane][r]

  // Q A-fragments (head dim K=0..31 / 32..63)
  Frag aq0, aq1;
  {
    const unsigned short* qr = qbase + nloc*HD;
    aq0.q4[0] = *(const uint4*)(qr +      half*8);
    aq0.q4[1] = *(const uint4*)(qr + 16 + half*8);
    aq1.q4[0] = *(const uint4*)(qr + 32 + half*8);
    aq1.q4[1] = *(const uint4*)(qr + 48 + half*8);
  }

  float mi[8], li[8];
  #pragma unroll
  for(int r=0;r<8;r++){ mi[r] = -1e30f; li[r] = 0.0f; }
  f32x8 co[4];
  #pragma unroll
  for(int i=0;i<4;i++) co[i] = zerov();

  int jmin = t0 - (WIN-1); if(jmin < 0) jmin = 0;
  const int j0s = jmin & ~31;
  unsigned short* Pw = P[wv];

  #pragma unroll 1
  for(int j0 = j0s; j0 <= t0 + 15; j0 += 32){
    // K^T fragments: [st 16-key subtile][dc head-dim chunk]
    Frag bk[4];
    #pragma unroll
    for(int st=0; st<2; st++){
      #pragma unroll
      for(int dc=0; dc<2; dc++){
        const unsigned* p = kbh + (((size_t)((j0 >> 4) + st)*2 + dc)*256) + lane*8;
        bk[st*2+dc].q4[0] = *(const uint4*)(p);
        bk[st*2+dc].q4[1] = *(const uint4*)(p + 4);
      }
    }
    f32x8 s0 = zerov(), s1 = zerov();
    s0 = wmma_bf16(aq0.v, bk[0].v, s0);
    s0 = wmma_bf16(aq1.v, bk[1].v, s0);
    s1 = wmma_bf16(aq0.v, bk[2].v, s1);
    s1 = wmma_bf16(aq1.v, bk[3].v, s1);

    // mask + online softmax (rows 0-7 in lanes<16, 8-15 in lanes>=16)
    #pragma unroll
    for(int r=0;r<8;r++){
      int qi = t0 + r + 8*half;
      int ja = j0 + nloc, jb = ja + 16;
      float sa = ((ja <= qi) && (ja > qi - WIN)) ? s0[r] : -1e30f;
      float sb = ((jb <= qi) && (jb > qi - WIN)) ? s1[r] : -1e30f;
      float mx = fmaxf(sa, sb);
      mx = fmaxf(mx, __shfl_xor(mx, 1, 16));
      mx = fmaxf(mx, __shfl_xor(mx, 2, 16));
      mx = fmaxf(mx, __shfl_xor(mx, 4, 16));
      mx = fmaxf(mx, __shfl_xor(mx, 8, 16));
      float nm = fmaxf(mi[r], mx);
      float al = __expf(mi[r] - nm);
      float pa = (sa > -1e29f) ? __expf(sa - nm) : 0.0f;
      float pb = (sb > -1e29f) ? __expf(sb - nm) : 0.0f;
      float rs = pa + pb;
      rs += __shfl_xor(rs, 1, 16);
      rs += __shfl_xor(rs, 2, 16);
      rs += __shfl_xor(rs, 4, 16);
      rs += __shfl_xor(rs, 8, 16);
      li[r] = li[r]*al + rs;
      mi[r] = nm;
      co[0][r] *= al; co[1][r] *= al; co[2][r] *= al; co[3][r] *= al;
      int mloc = r + 8*half;
      Pw[mloc*32 +      nloc] = f2bf(pa);
      Pw[mloc*32 + 16 + nloc] = f2bf(pb);
    }
    asm volatile("s_wait_dscnt 0" ::: "memory");   // LDS RAW before relayout

    // P: C-layout -> A-fragment via LDS
    Frag ap;
    #pragma unroll
    for(int p=0;p<4;p++){
      int kk = half*8 + 2*p;
      ap.u[p]   = *(const unsigned*)(Pw + nloc*32 +      kk);
      ap.u[4+p] = *(const unsigned*)(Pw + nloc*32 + 16 + kk);
    }

    // V fragments (pre-swizzled): P(16x32) @ V(32x64)
    const unsigned* vch = vbh + ((size_t)(j0 >> 5)*4)*256;
    #pragma unroll
    for(int ns=0; ns<4; ns++){
      const unsigned* p = vch + (size_t)ns*256 + lane*8;
      Frag bv;
      bv.q4[0] = *(const uint4*)(p);
      bv.q4[1] = *(const uint4*)(p + 4);
      co[ns] = wmma_bf16(ap.v, bv.v, co[ns]);
    }
  }

  // normalize + store attn_out[b][t][h*64+d] (row-major bf16 for GEMM2 A)
  #pragma unroll
  for(int r=0;r<8;r++){
    float inv = 1.0f / li[r];
    int t = t0 + r + 8*half;
    #pragma unroll
    for(int ns=0; ns<4; ns++){
      size_t o = (((size_t)b*T_SEQ + t)*NH + h)*HD + ns*16 + nloc;
      ao[o] = f2bf(co[ns][r] * inv);
    }
  }
}

// ---------------------------------------------------------------------------
// Kernel 3: out = attn_out(bf16) @ wout_sw, f32 to d_out. Wave = 32x64 tile.
// ---------------------------------------------------------------------------
__global__ __launch_bounds__(256)
void out_gemm_kernel(const unsigned short* __restrict__ ao, const unsigned* __restrict__ wsw,
                     float* __restrict__ out)
{
  const int lane = threadIdx.x & 31;
  const int wv   = threadIdx.x >> 5;
  const int wid  = blockIdx.x*8 + wv;      // 4096 waves = 256 mtiles * 16 ntiles
  const int mt   = wid & 255;
  const int nt64 = wid >> 8;
  const int mbase = mt*32;
  const int nloc = lane & 15;
  const int half = lane >> 4;

  f32x8 cc0[4], cc1[4];
  #pragma unroll
  for(int i=0;i<4;i++){ cc0[i] = zerov(); cc1[i] = zerov(); }

  const unsigned short* arow0 = ao + (size_t)(mbase +      nloc)*DM;
  const unsigned short* arow1 = ao + (size_t)(mbase + 16 + nloc)*DM;

  #pragma unroll 1
  for(int kc=0; kc<32; kc++){
    Frag a0, a1;
    a0.q4[0] = *(const uint4*)(arow0 + kc*32 +      half*8);
    a0.q4[1] = *(const uint4*)(arow0 + kc*32 + 16 + half*8);
    a1.q4[0] = *(const uint4*)(arow1 + kc*32 +      half*8);
    a1.q4[1] = *(const uint4*)(arow1 + kc*32 + 16 + half*8);
    if(kc+1 < 32)
      __builtin_prefetch(wsw + ((size_t)(kc+1)*64 + nt64*4)*256 + lane*8, 0, 1);
    #pragma unroll
    for(int ns=0; ns<4; ns++){
      const unsigned* bp = wsw + ((size_t)kc*64 + nt64*4 + ns)*256 + lane*8;
      Frag b;
      b.q4[0] = *(const uint4*)(bp);
      b.q4[1] = *(const uint4*)(bp + 4);
      cc0[ns] = wmma_bf16(a0.v, b.v, cc0[ns]);
      cc1[ns] = wmma_bf16(a1.v, b.v, cc1[ns]);
    }
  }

  #pragma unroll
  for(int hs=0; hs<2; hs++){
    #pragma unroll
    for(int ns=0; ns<4; ns++){
      #pragma unroll
      for(int r=0;r<8;r++){
        int m   = mbase + hs*16 + r + 8*half;
        int col = nt64*64 + ns*16 + nloc;
        out[(size_t)m*DM + col] = hs ? cc1[ns][r] : cc0[ns][r];
      }
    }
  }
}

// ---------------------------------------------------------------------------
extern "C" void kernel_launch(void* const* d_in, const int* in_sizes, int n_in,
                              void* d_out, int out_size, void* d_ws, size_t ws_size,
                              hipStream_t stream)
{
  const float* x    = (const float*)d_in[0];   // (4,2048,1024)
  const float* wqkv = (const float*)d_in[1];   // (1024,3072)
  const float* wout = (const float*)d_in[2];   // (1024,1024)
  float* out = (float*)d_out;                  // (4,2048,1024) f32

  char* ws = (char*)d_ws;
  const size_t MB = 1024*1024;
  unsigned short* xb   = (unsigned short*)(ws);              // 16 MB bf16 x
  unsigned short* attn = (unsigned short*)(ws + 16*MB);      // 16 MB bf16 attn_out
  unsigned short* qb   = (unsigned short*)(ws + 32*MB);      // 16 MB bf16 q [B][H][T][HD]
  unsigned*       kswz = (unsigned*)      (ws + 48*MB);      // 16 MB k fragments
  unsigned*       vswz = (unsigned*)      (ws + 64*MB);      // 16 MB v fragments
  unsigned*       wqsw = (unsigned*)      (ws + 80*MB);      // 6 MB swizzled w_qkv
  unsigned*       wosw = (unsigned*)      (ws + 88*MB);      // 2 MB swizzled w_out

  // prep
  cvt_x_kernel<<<2048, 256, 0, stream>>>((const float2*)x, (unsigned*)xb);
  swz_w_kernel<<<6144, 256, 0, stream>>>(wqkv, wqsw, 3*DM, 3*DM/16);  // 1.5M dwords
  swz_w_kernel<<<2048, 256, 0, stream>>>(wout, wosw, DM, DM/16);      // 0.5M dwords

  // qkv projection: 12288 waves (32x64 tiles)
  qkv_gemm_kernel<<<1536, 256, 0, stream>>>(xb, wqsw, qb, (unsigned short*)kswz,
                                            (unsigned short*)vswz);

  // attention: 8192 wave-tiles
  swa_kernel<<<1024, 256, 0, stream>>>(qb, kswz, vswz, attn);

  // output projection: 4096 waves (32x64 tiles)
  out_gemm_kernel<<<512, 256, 0, stream>>>(attn, wosw, out);
}
